// MyBert_V1_80272938762708
// MI455X (gfx1250) — compile-verified, matrix operand order checked by reading, not run on hardware
//
#include <hip/hip_runtime.h>
#include <hip/hip_bf16.h>

// ---------------- constants from the reference ----------------
constexpr int CB = 16, CS = 256, CD = 768, CL = 12, CNH = 12, CF = 3072, CV = 21128;
constexpr int CNE = 30000, CNM = 15000, CNF = 16;
constexpr int CEB = 50000, CEC = 50000, CER = 30000;
constexpr int CLD = 2048, CAH = 8, CRED = 256;

// ---------------- WMMA types ----------------
typedef __attribute__((ext_vector_type(16))) __bf16 v16bf;
typedef __attribute__((ext_vector_type(8)))  float  v8f;
typedef __attribute__((ext_vector_type(8)))  unsigned int v8u;

__device__ inline unsigned short f2bf(float f) {
    unsigned int u = __builtin_bit_cast(unsigned int, f);
    unsigned int r = u + 0x7fffu + ((u >> 16) & 1u);   // round-to-nearest-even
    return (unsigned short)(r >> 16);
}
__device__ inline unsigned int pack2bf(float a, float b) {
    return (unsigned int)f2bf(a) | ((unsigned int)f2bf(b) << 16);
}

// ordered-uint mapping for float atomic max
__device__ inline unsigned int ford(float f) {
    unsigned int u = __builtin_bit_cast(unsigned int, f);
    return (u & 0x80000000u) ? ~u : (u | 0x80000000u);
}
__device__ inline float funord(unsigned int u) {
    u = (u & 0x80000000u) ? (u & 0x7fffffffu) : ~u;
    return __builtin_bit_cast(float, u);
}

// ============================================================
// Generic batched bf16-WMMA GEMM:  C = act(alpha * A@B(^T) + bias)
// block tile 64(M) x 128(N), BK=32, 8 waves (2x4), 2x2 WMMA tiles/wave.
// Interior tiles take a branchless b128-load / b64-LDS-store fast path;
// only edge tiles / K remainders run the guarded scalar path.
// ============================================================
#define BM 64
#define BN 128
#define BK 32
#define LPAD 4   // row stride BK+4 halves = 72B: 8B aligned rows, 4B aligned K-pairs

__global__ void __launch_bounds__(256)
gemm_bf16_wmma(const float* __restrict__ A, const float* __restrict__ Bm,
               const float* __restrict__ bias, float* __restrict__ C,
               int M, int N, int K, int lda, int ldb, int ldc,
               int zdiv,
               long long sA1, long long sA2, long long sB1, long long sB2,
               long long sC1, long long sC2, long long sb1, long long sb2,
               int transB, float alpha, int act)
{
    __shared__ __align__(16) unsigned short sAt[BM][BK + LPAD];
    __shared__ __align__(16) unsigned short sBt[BN][BK + LPAD];

    int bz = blockIdx.z;
    int z1 = bz / zdiv, z2 = bz % zdiv;
    A  += z1 * sA1 + z2 * sA2;
    Bm += z1 * sB1 + z2 * sB2;
    C  += z1 * sC1 + z2 * sC2;
    if (bias) bias += z1 * sb1 + z2 * sb2;

    int row0 = blockIdx.y * BM;
    int col0 = blockIdx.x * BN;
    int tid  = threadIdx.x;
    int lane = tid & 31;
    int wave = tid >> 5;
    int wm = wave & 1;   // 2 waves along M
    int wn = wave >> 1;  // 4 waves along N

    const bool fullTile = (row0 + BM <= M) && (col0 + BN <= N);

    v8f acc[2][2];
    for (int i = 0; i < 2; ++i)
        for (int j = 0; j < 2; ++j) acc[i][j] = 0.f;

    for (int k0 = 0; k0 < K; k0 += BK) {
        bool fast = fullTile && (k0 + BK <= K);
        if (fast) {
            // ---- A tile: 64x32 floats as 512 float4 quads ----
#pragma unroll
            for (int i = 0; i < 2; ++i) {
                int q = tid + i * 256;
                int m = q >> 3, kq = (q & 7) * 4;
                float4 v = *(const float4*)(A + (long long)(row0 + m) * lda + k0 + kq);
                unsigned long long pv = (unsigned long long)pack2bf(v.x, v.y)
                                      | ((unsigned long long)pack2bf(v.z, v.w) << 32);
                *(unsigned long long*)&sAt[m][kq] = pv;
            }
            // ---- B tile -> sBt[n][k] ----
            if (transB) {   // B stored [N,K]: contiguous along k
#pragma unroll
                for (int i = 0; i < 4; ++i) {
                    int q = tid + i * 256;
                    int n = q >> 3, kq = (q & 7) * 4;
                    float4 v = *(const float4*)(Bm + (long long)(col0 + n) * ldb + k0 + kq);
                    unsigned long long pv = (unsigned long long)pack2bf(v.x, v.y)
                                          | ((unsigned long long)pack2bf(v.z, v.w) << 32);
                    *(unsigned long long*)&sBt[n][kq] = pv;
                }
            } else {        // B stored [K,N]: coalesced along n, transpose into LDS
#pragma unroll
                for (int i = 0; i < 4; ++i) {
                    int q = tid + i * 256;
                    int k = q >> 5, nq = (q & 31) * 4;
                    float4 v = *(const float4*)(Bm + (long long)(k0 + k) * ldb + col0 + nq);
                    sBt[nq + 0][k] = f2bf(v.x);
                    sBt[nq + 1][k] = f2bf(v.y);
                    sBt[nq + 2][k] = f2bf(v.z);
                    sBt[nq + 3][k] = f2bf(v.w);
                }
            }
        } else {
            // ---- guarded scalar staging (edge tiles / K remainder) ----
#pragma unroll
            for (int i = 0; i < 8; ++i) {
                int e = tid + i * 256;
                int m = e >> 5, k = e & 31;
                int gm = row0 + m, gk = k0 + k;
                float v = (gm < M && gk < K) ? A[(long long)gm * lda + gk] : 0.f;
                sAt[m][k] = f2bf(v);
            }
#pragma unroll
            for (int i = 0; i < 16; ++i) {
                int e = tid + i * 256;
                int n = e >> 5, k = e & 31;
                int gn = col0 + n, gk = k0 + k;
                float v = 0.f;
                if (gn < N && gk < K)
                    v = transB ? Bm[(long long)gn * ldb + gk]
                               : Bm[(long long)gk * ldb + gn];
                sBt[n][k] = f2bf(v);
            }
        }
        __syncthreads();

        int khalf = lane >> 4;   // lanes 0-15 -> low K halves, 16-31 -> high
        int lidx  = lane & 15;
        v8u au[2], bu[2];
#pragma unroll
        for (int t = 0; t < 2; ++t) {
            const unsigned short* ap = &sAt[wm * 32 + t * 16 + lidx][0];
            const unsigned short* bp = &sBt[wn * 32 + t * 16 + lidx][0];
#pragma unroll
            for (int j = 0; j < 8; ++j) {
                // 16-bit A-matrix layout: VGPR j<4 -> K = khalf*8 + 2j (+1)
                //                         VGPR j>=4 -> K = 16 + khalf*8 + 2(j-4)
                int kp = ((j & 4) ? 16 : 0) + khalf * 8 + (j & 3) * 2;
                au[t][j] = *(const unsigned int*)(ap + kp);
                bu[t][j] = *(const unsigned int*)(bp + kp);
            }
        }
#pragma unroll
        for (int tm = 0; tm < 2; ++tm)
#pragma unroll
            for (int tn = 0; tn < 2; ++tn) {
                v16bf av = __builtin_bit_cast(v16bf, au[tm]);
                v16bf bv = __builtin_bit_cast(v16bf, bu[tn]);
                acc[tm][tn] = __builtin_amdgcn_wmma_f32_16x16x32_bf16(
                    false, av, false, bv, (short)0, acc[tm][tn], false, false);
            }
        __syncthreads();
    }

    // --- epilogue: C/D layout: lane<16 -> M=r, lane>=16 -> M=r+8; N=lane&15 ---
    int lidx = lane & 15, khalf = lane >> 4;
#pragma unroll
    for (int tm = 0; tm < 2; ++tm)
#pragma unroll
        for (int tn = 0; tn < 2; ++tn) {
            int c = col0 + wn * 32 + tn * 16 + lidx;
            if (!fullTile && c >= N) continue;
            float bv = bias ? bias[c] : 0.f;
#pragma unroll
            for (int r = 0; r < 8; ++r) {
                int m = row0 + wm * 32 + tm * 16 + khalf * 8 + r;
                if (!fullTile && m >= M) continue;
                float v = alpha * acc[tm][tn][r] + bv;
                if      (act == 1) v = 0.5f * v * (1.f + erff(v * 0.7071067811865476f)); // gelu exact
                else if (act == 2) v = tanhf(v);
                else if (act == 3) v = v > 0.f ? v : 0.01f * v;  // leaky 0.01
                else if (act == 4) v = v > 0.f ? v : 0.2f  * v;  // leaky 0.2
                C[(long long)m * ldc + c] = v;
            }
        }
}

// ============================================================
// elementwise / reduction helpers
// ============================================================
__global__ void embed_kernel(const int* __restrict__ ids, const float* __restrict__ we,
                             const float* __restrict__ pe, const float* __restrict__ te,
                             float* __restrict__ x)
{
    int tok = blockIdx.x;           // B*S
    int s = tok % CS;
    int id = ids[tok];
    for (int d = threadIdx.x; d < CD; d += blockDim.x)
        x[(long long)tok * CD + d] = we[(long long)id * CD + d] + pe[s * CD + d] + te[d];
}

__global__ void __launch_bounds__(256)
ln_kernel(const float* __restrict__ x, const float* __restrict__ res,
          const float* __restrict__ g, const float* __restrict__ b,
          float* __restrict__ y, int Dn)
{
    long long t = blockIdx.x;
    __shared__ float red[256];
    __shared__ float s_mean, s_rstd;
    int tid = threadIdx.x;
    float ls = 0.f;
    for (int d = tid; d < Dn; d += 256) {
        float v = x[t * Dn + d] + (res ? res[t * Dn + d] : 0.f);
        ls += v;
    }
    red[tid] = ls; __syncthreads();
    for (int s2 = 128; s2 > 0; s2 >>= 1) { if (tid < s2) red[tid] += red[tid + s2]; __syncthreads(); }
    if (tid == 0) s_mean = red[0] / Dn;
    __syncthreads();
    float mean = s_mean;
    float lv = 0.f;
    for (int d = tid; d < Dn; d += 256) {
        float v = x[t * Dn + d] + (res ? res[t * Dn + d] : 0.f) - mean;
        lv += v * v;
    }
    red[tid] = lv; __syncthreads();
    for (int s2 = 128; s2 > 0; s2 >>= 1) { if (tid < s2) red[tid] += red[tid + s2]; __syncthreads(); }
    if (tid == 0) s_rstd = rsqrtf(red[0] / Dn + 1e-12f);
    __syncthreads();
    float rstd = s_rstd;
    for (int d = tid; d < Dn; d += 256) {
        float v = x[t * Dn + d] + (res ? res[t * Dn + d] : 0.f);
        y[t * Dn + d] = (v - mean) * rstd * g[d] + b[d];
    }
}

// softmax over S=256 with attention mask; one row per block (blockDim = 256)
__global__ void __launch_bounds__(256)
softmax_mask_kernel(float* __restrict__ sc, const float* __restrict__ mask)
{
    long long row = blockIdx.x;                 // (b*NH+h)*S + s
    int b = (int)(row / ((long long)CNH * CS));
    float* p = sc + row * CS;
    int t = threadIdx.x;
    __shared__ float red[256];
    float v = p[t] + (mask[b * CS + t] - 1.f) * 1e9f;
    red[t] = v; __syncthreads();
    for (int s2 = 128; s2 > 0; s2 >>= 1) { if (t < s2) red[t] = fmaxf(red[t], red[t + s2]); __syncthreads(); }
    float mx = red[0]; __syncthreads();
    float e = __expf(v - mx);
    red[t] = e; __syncthreads();
    for (int s2 = 128; s2 > 0; s2 >>= 1) { if (t < s2) red[t] += red[t + s2]; __syncthreads(); }
    p[t] = e / red[0];
}

__global__ void meanpool_kernel(const float* __restrict__ x, float* __restrict__ out)
{
    int b = blockIdx.x;
    int d = blockIdx.y * 256 + threadIdx.x;
    if (d >= CD) return;
    float s = 0.f;
    for (int i = 0; i < CS; ++i) s += x[((long long)b * CS + i) * CD + d];
    out[b * CD + d] = s * (1.f / CS);
}

__global__ void fill_f32(float* p, float v, long long n)
{ long long i = (long long)blockIdx.x * 256 + threadIdx.x; if (i < n) p[i] = v; }

__global__ void fill_u32(unsigned int* p, unsigned int v, long long n)
{ long long i = (long long)blockIdx.x * 256 + threadIdx.x; if (i < n) p[i] = v; }

__global__ void copy2d_kernel(const float* __restrict__ src, int ls,
                              float* __restrict__ dst, int ld, int M, int N)
{
    long long i = (long long)blockIdx.x * 256 + threadIdx.x;
    if (i >= (long long)M * N) return;
    int m = (int)(i / N), n = (int)(i % N);
    dst[(long long)m * ld + n] = src[(long long)m * ls + n];
}

__global__ void addbias2_kernel(float* __restrict__ out, const float* __restrict__ b1,
                                const float* __restrict__ b2, long long n, int C)
{
    long long i = (long long)blockIdx.x * 256 + threadIdx.x;
    if (i >= n) return;
    int c = (int)(i % C);
    out[i] += b1[c] + (b2 ? b2[c] : 0.f);
}

// ============================================================
// GAT helpers
// ============================================================
// P[k*H+h] = sum_c W[k*H*C + h*C + c] * a[h*C + c]
__global__ void wa_kernel(const float* __restrict__ W, const float* __restrict__ a,
                          float* __restrict__ P, int K, int C, int H)
{
    int idx = blockIdx.x * blockDim.x + threadIdx.x;
    if (idx >= K * H) return;
    int k = idx / H, h = idx % H;
    const float* wp = W + (long long)k * H * C + (long long)h * C;
    const float* ap = a + (long long)h * C;
    float s = 0.f;
    for (int c = 0; c < C; ++c) s += wp[c] * ap[c];
    P[idx] = s;
}

// out[n*H+h] = sum_k X[n*K+k] * P[k*H+h]   (one wave32 per (n,h))
__global__ void __launch_bounds__(256)
al_kernel(const float* __restrict__ X, const float* __restrict__ P,
          float* __restrict__ out, int Nn, int K, int H)
{
    int gw = (blockIdx.x * 256 + threadIdx.x) >> 5;
    int lane = threadIdx.x & 31;
    if (gw >= Nn * H) return;
    int n = gw / H, h = gw % H;
    const float* xp = X + (long long)n * K;
    float s = 0.f;
    for (int k = lane; k < K; k += 32) s += xp[k] * P[k * H + h];
    for (int m = 16; m > 0; m >>= 1) s += __shfl_xor(s, m, 32);
    if (lane == 0) out[gw] = s;
}

__global__ void edge_max_kernel(const float* __restrict__ als, const float* __restrict__ ald,
                                const int* __restrict__ src, const int* __restrict__ dst,
                                float* __restrict__ e, unsigned int* __restrict__ mx,
                                int E, int H)
{
    int i = blockIdx.x * blockDim.x + threadIdx.x;
    if (i >= E * H) return;
    int ed = i / H, h = i % H;
    float v = als[src[ed] * H + h] + ald[dst[ed] * H + h];
    v = v > 0.f ? v : 0.2f * v;          // leaky_relu 0.2
    e[i] = v;
    atomicMax(&mx[dst[ed] * H + h], ford(v));
}

__global__ void edge_sum_kernel(const float* __restrict__ e, const int* __restrict__ dst,
                                const unsigned int* __restrict__ mx, float* __restrict__ z,
                                int E, int H)
{
    int i = blockIdx.x * blockDim.x + threadIdx.x;
    if (i >= E * H) return;
    int ed = i / H, h = i % H;
    float m = funord(mx[dst[ed] * H + h]);
    atomicAdd(&z[dst[ed] * H + h], __expf(e[i] - m));
}

__global__ void edge_scatter_kernel(const float* __restrict__ e, const int* __restrict__ src,
                                    const int* __restrict__ dst, const unsigned int* __restrict__ mx,
                                    const float* __restrict__ z, const float* __restrict__ hs,
                                    float* __restrict__ out, int E, int H, int C)
{
    int ed = blockIdx.x;
    int f = blockIdx.y * blockDim.x + threadIdx.x;
    if (f >= H * C) return;
    int h = f / C;
    int di = dst[ed];
    float m = funord(mx[di * H + h]);
    float p = __expf(e[(long long)ed * H + h] - m);
    float alpha = p / (z[di * H + h] + 1e-16f);
    atomicAdd(&out[(long long)di * H * C + f],
              hs[(long long)src[ed] * H * C + f] * alpha);
}

// ============================================================
// fusion attention: 3 tokens, 8 heads, dh=256; one block per (b,h)
// ============================================================
__global__ void __launch_bounds__(256)
fuse_attn_kernel(const float* __restrict__ qh, const float* __restrict__ kh,
                 const float* __restrict__ vh, float* __restrict__ ctx)
{
    int bh = blockIdx.x;
    int b = bh / CAH, h = bh % CAH;
    int d = threadIdx.x;
    __shared__ float red[256];
    __shared__ float sc[3][3];
    float qv[3], kv[3], vv[3];
    for (int s = 0; s < 3; ++s) {
        long long off = ((long long)b * 3 + s) * CLD + h * 256 + d;
        qv[s] = qh[off]; kv[s] = kh[off]; vv[s] = vh[off];
    }
    for (int s = 0; s < 3; ++s)
        for (int t = 0; t < 3; ++t) {
            red[d] = qv[s] * kv[t]; __syncthreads();
            for (int m = 128; m > 0; m >>= 1) { if (d < m) red[d] += red[d + m]; __syncthreads(); }
            if (d == 0) sc[s][t] = red[0] * (1.f / 16.f);   // 1/sqrt(256)
            __syncthreads();
        }
    float a[3][3];
    for (int s = 0; s < 3; ++s) {
        float mx = fmaxf(sc[s][0], fmaxf(sc[s][1], sc[s][2]));
        float e0 = __expf(sc[s][0] - mx), e1 = __expf(sc[s][1] - mx), e2 = __expf(sc[s][2] - mx);
        float zz = e0 + e1 + e2;
        a[s][0] = e0 / zz; a[s][1] = e1 / zz; a[s][2] = e2 / zz;
    }
    for (int s = 0; s < 3; ++s) {
        float o = a[s][0] * vv[0] + a[s][1] * vv[1] + a[s][2] * vv[2];
        ctx[((long long)b * 3 + s) * CLD + h * 256 + d] = o;
    }
}

// ============================================================
// host orchestration
// ============================================================
extern "C" void kernel_launch(void* const* d_in, const int* in_sizes, int n_in,
                              void* d_out, int out_size, void* d_ws, size_t ws_size,
                              hipStream_t stream)
{
    (void)in_sizes; (void)n_in; (void)out_size; (void)ws_size;

    const int*   input_ids  = (const int*)  d_in[0];
    const float* attn_mask  = (const float*)d_in[1];
    const float* ent_x      = (const float*)d_in[2];
    const float* msg_x      = (const float*)d_in[3];
    const float* father_x   = (const float*)d_in[4];
    /* d_in[5], d_in[6] = has_src/has_dst : 'msg1' is dead in the reference -> skipped */
    const int*   belong_src = (const int*)  d_in[7];
    const int*   belong_dst = (const int*)  d_in[8];
    const int*   co_src     = (const int*)  d_in[9];
    const int*   co_dst     = (const int*)  d_in[10];
    const int*   refer_src  = (const int*)  d_in[11];
    const int*   refer_dst  = (const int*)  d_in[12];
    const float* logits_diff= (const float*)d_in[13];
    const float* bp         = (const float*)d_in[14];
    const float* gp         = (const float*)d_in[15];
    const float* fp         = (const float*)d_in[16];

    // ---- flat offsets (insertion order) into param blobs ----
    size_t off = 0;
    auto take = [&](size_t n) { size_t r = off; off += n; return r; };
    const size_t o_word = take((size_t)CV * CD), o_pos = take((size_t)CS * CD),
                 o_type = take(CD), o_eg = take(CD), o_eb = take(CD),
                 o_Wq = take((size_t)CL * CD * CD), o_bq = take((size_t)CL * CD),
                 o_Wk = take((size_t)CL * CD * CD), o_bk = take((size_t)CL * CD),
                 o_Wv = take((size_t)CL * CD * CD), o_bv = take((size_t)CL * CD),
                 o_Wo = take((size_t)CL * CD * CD), o_bo = take((size_t)CL * CD),
                 o_g1 = take((size_t)CL * CD), o_c1 = take((size_t)CL * CD),
                 o_W1 = take((size_t)CL * CD * CF), o_b1 = take((size_t)CL * CF),
                 o_W2 = take((size_t)CL * CF * CD), o_b2 = take((size_t)CL * CD),
                 o_g2 = take((size_t)CL * CD), o_c2 = take((size_t)CL * CD),
                 o_pw = take((size_t)CD * CD), o_pb = take(CD);
    off = 0;
    const size_t go_w1 = take((size_t)4 * 300 * 3072), go_as = take((size_t)4 * 4 * 768),
                 go_ad = take((size_t)4 * 4 * 768), go_b1 = take((size_t)4 * 3072),
                 go_w2 = take((size_t)4 * 3072 * 768), go_as2 = take((size_t)4 * 768),
                 go_ad2 = take((size_t)4 * 768), go_b2 = take((size_t)4 * 768);
    off = 0;
    const size_t fo_qlw = take((size_t)CLD * CLD), fo_qlb = take(CLD),
                 fo_klw = take((size_t)CLD * CLD), fo_klb = take(CLD),
                 fo_vlw = take((size_t)CLD * CLD), fo_vlb = take(CLD),
                 fo_inw = take((size_t)3 * CLD * CLD), fo_inb = take((size_t)3 * CLD),
                 fo_outw = take((size_t)CLD * CLD), fo_outb = take(CLD),
                 fo_redw = take((size_t)3 * CLD * CRED), fo_redb = take((size_t)3 * CRED),
                 fo_ldw = take((size_t)3 * CRED * CRED), fo_ldb = take(CRED),
                 fo_m1w = take((size_t)1792 * 896), fo_m1b = take(896),
                 fo_m2w = take((size_t)896 * 2), fo_m2b = take(2);

    // ---- workspace bump allocator ----
    char* wsb = (char*)d_ws;
    size_t wo = 0;
    auto alloc = [&](size_t nfl) -> float* {
        float* p = (float*)(wsb + wo);
        wo += ((nfl * 4 + 255) / 256) * 256;
        return p;
    };

    auto gemm = [&](const float* A, int lda, const float* Bw, int ldb,
                    const float* bias, float* Cp, int ldc,
                    int M, int N, int K, int transB, float alpha, int act,
                    int batch = 1, int zdiv = 1,
                    long long sA1 = 0, long long sA2 = 0,
                    long long sB1 = 0, long long sB2 = 0,
                    long long sC1 = 0, long long sC2 = 0,
                    long long sb1 = 0, long long sb2 = 0) {
        dim3 g((N + BN - 1) / BN, (M + BM - 1) / BM, batch);
        gemm_bf16_wmma<<<g, dim3(256), 0, stream>>>(A, Bw, bias, Cp, M, N, K, lda, ldb, ldc,
                                                    zdiv, sA1, sA2, sB1, sB2, sC1, sC2,
                                                    sb1, sb2, transB, alpha, act);
    };
    auto fillf = [&](float* p, float v, long long n) {
        fill_f32<<<dim3((unsigned)((n + 255) / 256)), 256, 0, stream>>>(p, v, n);
    };
    auto fillu = [&](unsigned int* p, unsigned int v, long long n) {
        fill_u32<<<dim3((unsigned)((n + 255) / 256)), 256, 0, stream>>>(p, v, n);
    };
    auto copy2d = [&](const float* s, int ls, float* dd, int ld, int M, int N) {
        long long n = (long long)M * N;
        copy2d_kernel<<<dim3((unsigned)((n + 255) / 256)), 256, 0, stream>>>(s, ls, dd, ld, M, N);
    };

    // =================== BERT ===================
    const int T = CB * CS;
    float* xbuf  = alloc((size_t)T * CD);
    float* x1buf = alloc((size_t)T * CD);
    float* tmp   = alloc((size_t)T * CD);
    float* qb    = alloc((size_t)T * CD);
    float* kb    = alloc((size_t)T * CD);
    float* vb    = alloc((size_t)T * CD);
    float* scores= alloc((size_t)CB * CNH * CS * CS);
    float* ctxb  = alloc((size_t)T * CD);
    float* hb    = alloc((size_t)T * CF);

    embed_kernel<<<T, 256, 0, stream>>>(input_ids, bp + o_word, bp + o_pos, bp + o_type, tmp);
    ln_kernel<<<T, 256, 0, stream>>>(tmp, nullptr, bp + o_eg, bp + o_eb, xbuf, CD);

    for (int l = 0; l < CL; ++l) {
        const float* Wq = bp + o_Wq + (size_t)l * CD * CD;
        const float* Wk = bp + o_Wk + (size_t)l * CD * CD;
        const float* Wv = bp + o_Wv + (size_t)l * CD * CD;
        const float* Wo = bp + o_Wo + (size_t)l * CD * CD;
        gemm(xbuf, CD, Wq, CD, bp + o_bq + (size_t)l * CD, qb, CD, T, CD, CD, 0, 1.f, 0);
        gemm(xbuf, CD, Wk, CD, bp + o_bk + (size_t)l * CD, kb, CD, T, CD, CD, 0, 1.f, 0);
        gemm(xbuf, CD, Wv, CD, bp + o_bv + (size_t)l * CD, vb, CD, T, CD, CD, 0, 1.f, 0);
        // scores[b,h] = q @ k^T / 8    (per-(b,h) strided batch)
        gemm(qb, CD, kb, CD, nullptr, scores, CS, CS, CS, 64, 1, 0.125f, 0,
             CB * CNH, CNH,
             (long long)CS * CD, 64, (long long)CS * CD, 64,
             (long long)CNH * CS * CS, (long long)CS * CS);
        softmax_mask_kernel<<<CB * CNH * CS, 256, 0, stream>>>(scores, attn_mask);
        // ctx[b,h] = a @ v  -> written interleaved as [B,S,D]
        gemm(scores, CS, vb, CD, nullptr, ctxb, CD, CS, 64, CS, 0, 1.f, 0,
             CB * CNH, CNH,
             (long long)CNH * CS * CS, (long long)CS * CS,
             (long long)CS * CD, 64, (long long)CS * CD, 64);
        gemm(ctxb, CD, Wo, CD, bp + o_bo + (size_t)l * CD, tmp, CD, T, CD, CD, 0, 1.f, 0);
        ln_kernel<<<T, 256, 0, stream>>>(tmp, xbuf, bp + o_g1 + (size_t)l * CD,
                                         bp + o_c1 + (size_t)l * CD, x1buf, CD);
        gemm(x1buf, CD, bp + o_W1 + (size_t)l * CD * CF, CF,
             bp + o_b1 + (size_t)l * CF, hb, CF, T, CF, CD, 0, 1.f, 1 /*gelu*/);
        gemm(hb, CF, bp + o_W2 + (size_t)l * CF * CD, CD,
             bp + o_b2 + (size_t)l * CD, tmp, CD, T, CD, CF, 0, 1.f, 0);
        ln_kernel<<<T, 256, 0, stream>>>(tmp, x1buf, bp + o_g2 + (size_t)l * CD,
                                         bp + o_c2 + (size_t)l * CD, xbuf, CD);
    }

    float* poolt  = alloc((size_t)T * CD);
    float* pooled = alloc((size_t)CB * CD);
    gemm(xbuf, CD, bp + o_pw, CD, bp + o_pb, poolt, CD, T, CD, CD, 0, 1.f, 2 /*tanh*/);
    meanpool_kernel<<<dim3(CB, 3), 256, 0, stream>>>(poolt, pooled);

    // =================== GAT ===================
    const unsigned int NEG_INF_ORD = 0x007FFFFFu;   // ford(-inf)
    float* Ps1 = alloc(300 * 4); float* Pd1 = alloc(300 * 4);
    float* Ps2 = alloc(300 * 4); float* Pd2 = alloc(300 * 4);
    float* Ps3 = alloc(300 * 4); float* Pd3 = alloc(300 * 4);
    wa_kernel<<<(300 * 4 + 255) / 256, 256, 0, stream>>>(gp + go_w1 + (size_t)1 * 300 * 3072, gp + go_as + 1 * 4 * 768, Ps1, 300, 768, 4);
    wa_kernel<<<(300 * 4 + 255) / 256, 256, 0, stream>>>(gp + go_w1 + (size_t)1 * 300 * 3072, gp + go_ad + 1 * 4 * 768, Pd1, 300, 768, 4);
    wa_kernel<<<(300 * 4 + 255) / 256, 256, 0, stream>>>(gp + go_w1 + (size_t)2 * 300 * 3072, gp + go_as + 2 * 4 * 768, Ps2, 300, 768, 4);
    wa_kernel<<<(300 * 4 + 255) / 256, 256, 0, stream>>>(gp + go_w1 + (size_t)2 * 300 * 3072, gp + go_ad + 2 * 4 * 768, Pd2, 300, 768, 4);
    wa_kernel<<<(300 * 4 + 255) / 256, 256, 0, stream>>>(gp + go_w1 + (size_t)3 * 300 * 3072, gp + go_as + 3 * 4 * 768, Ps3, 300, 768, 4);
    wa_kernel<<<(300 * 4 + 255) / 256, 256, 0, stream>>>(gp + go_w1 + (size_t)3 * 300 * 3072, gp + go_ad + 3 * 4 * 768, Pd3, 300, 768, 4);

    float* al_s1 = alloc((size_t)CNM * 4); float* al_d1 = alloc((size_t)CNE * 4);
    float* al_s2 = alloc((size_t)CNE * 4); float* al_d2 = alloc((size_t)CNE * 4);
    float* al_s3 = alloc((size_t)CNE * 4); float* al_d3 = alloc((size_t)CNF * 4);
    auto al = [&](const float* X, const float* P, float* o, int Nn, int K, int H) {
        int waves = Nn * H;
        al_kernel<<<(waves + 7) / 8, 256, 0, stream>>>(X, P, o, Nn, K, H);
    };
    al(msg_x, Ps1, al_s1, CNM, 300, 4);
    al(ent_x, Pd1, al_d1, CNE, 300, 4);
    al(ent_x, Ps2, al_s2, CNE, 300, 4);
    al(ent_x, Pd2, al_d2, CNE, 300, 4);
    al(ent_x, Ps3, al_s3, CNE, 300, 4);
    al(father_x, Pd3, al_d3, CNF, 300, 4);

    float* hs1 = alloc((size_t)CNM * 3072);
    float* hs2 = alloc((size_t)CNE * 3072);
    float* hs3 = alloc((size_t)CNE * 3072);
    gemm(msg_x, 300, gp + go_w1 + (size_t)1 * 300 * 3072, 3072, nullptr, hs1, 3072, CNM, 3072, 300, 0, 1.f, 0);
    gemm(ent_x, 300, gp + go_w1 + (size_t)2 * 300 * 3072, 3072, nullptr, hs2, 3072, CNE, 3072, 300, 0, 1.f, 0);
    gemm(ent_x, 300, gp + go_w1 + (size_t)3 * 300 * 3072, 3072, nullptr, hs3, 3072, CNE, 3072, 300, 0, 1.f, 0);

    float* ent1 = alloc((size_t)CNE * 3072);
    float* fat1 = alloc((size_t)CNF * 3072);
    fillf(ent1, 0.f, (long long)CNE * 3072);
    fillf(fat1, 0.f, (long long)CNF * 3072);

    unsigned int* mxb = (unsigned int*)alloc((size_t)CNE * 4);
    float* zb = alloc((size_t)CNE * 4);
    float* eb = alloc((size_t)CEB * 4);

    auto gat_scatter = [&](const float* als, const float* ald, const int* src, const int* dst,
                           const float* hs, float* out, int E, int nd, int H, int C) {
        fillu(mxb, NEG_INF_ORD, (long long)nd * H);
        fillf(zb, 0.f, (long long)nd * H);
        int n = E * H;
        edge_max_kernel<<<(n + 255) / 256, 256, 0, stream>>>(als, ald, src, dst, eb, mxb, E, H);
        edge_sum_kernel<<<(n + 255) / 256, 256, 0, stream>>>(eb, dst, mxb, zb, E, H);
        edge_scatter_kernel<<<dim3(E, (H * C) / 256), 256, 0, stream>>>(eb, src, dst, mxb, zb, hs, out, E, H, C);
    };
    gat_scatter(al_s1, al_d1, belong_src, belong_dst, hs1, ent1, CEB, CNE, 4, 768);  // msg -> ent
    gat_scatter(al_s2, al_d2, co_src,     co_dst,     hs2, ent1, CEC, CNE, 4, 768);  // ent -> ent
    gat_scatter(al_s3, al_d3, refer_src,  refer_dst,  hs3, fat1, CER, CNF, 4, 768);  // ent -> father
    addbias2_kernel<<<(unsigned)(((long long)CNE * 3072 + 255) / 256), 256, 0, stream>>>(
        ent1, gp + go_b1 + 1 * 3072, gp + go_b1 + 2 * 3072, (long long)CNE * 3072, 3072);
    addbias2_kernel<<<(unsigned)(((long long)CNF * 3072 + 255) / 256), 256, 0, stream>>>(
        fat1, gp + go_b1 + 3 * 3072, nullptr, (long long)CNF * 3072, 3072);

    // layer 2 ('refer' only, heads=1, out 768)
    float* P2s = alloc(3072); float* P2d = alloc(3072);
    wa_kernel<<<(3072 + 255) / 256, 256, 0, stream>>>(gp + go_w2 + (size_t)3 * 3072 * 768, gp + go_as2 + 3 * 768, P2s, 3072, 768, 1);
    wa_kernel<<<(3072 + 255) / 256, 256, 0, stream>>>(gp + go_w2 + (size_t)3 * 3072 * 768, gp + go_ad2 + 3 * 768, P2d, 3072, 768, 1);
    float* al2s = alloc(CNE); float* al2d = alloc(CNF);
    al(ent1, P2s, al2s, CNE, 3072, 1);
    al(fat1, P2d, al2d, CNF, 3072, 1);
    float* hsL2 = alloc((size_t)CNE * 768);
    gemm(ent1, 3072, gp + go_w2 + (size_t)3 * 3072 * 768, 768, nullptr, hsL2, 768, CNE, 768, 3072, 0, 1.f, 0);
    float* fat2 = alloc((size_t)CNF * 768);
    fillf(fat2, 0.f, (long long)CNF * 768);
    gat_scatter(al2s, al2d, refer_src, refer_dst, hsL2, fat2, CER, CNF, 1, 768);
    addbias2_kernel<<<(unsigned)(((long long)CNF * 768 + 255) / 256), 256, 0, stream>>>(
        fat2, gp + go_b2 + 3 * 768, nullptr, (long long)CNF * 768, 768);

    // =================== fusion module ===================
    const int M3 = CB * 3;   // 48
    float* qf = alloc((size_t)M3 * CLD); float* kf = alloc((size_t)M3 * CLD); float* vf = alloc((size_t)M3 * CLD);
    gemm(logits_diff, CLD, fp + fo_qlw, CLD, fp + fo_qlb, qf, CLD, M3, CLD, CLD, 0, 1.f, 0);
    gemm(logits_diff, CLD, fp + fo_klw, CLD, fp + fo_klb, kf, CLD, M3, CLD, CLD, 0, 1.f, 0);
    gemm(logits_diff, CLD, fp + fo_vlw, CLD, fp + fo_vlb, vf, CLD, M3, CLD, CLD, 0, 1.f, 0);
    float* qh = alloc((size_t)M3 * CLD); float* kh = alloc((size_t)M3 * CLD); float* vh = alloc((size_t)M3 * CLD);
    gemm(qf, CLD, fp + fo_inw + (size_t)0 * CLD * CLD, CLD, fp + fo_inb + 0 * CLD, qh, CLD, M3, CLD, CLD, 0, 1.f, 0);
    gemm(kf, CLD, fp + fo_inw + (size_t)1 * CLD * CLD, CLD, fp + fo_inb + 1 * CLD, kh, CLD, M3, CLD, CLD, 0, 1.f, 0);
    gemm(vf, CLD, fp + fo_inw + (size_t)2 * CLD * CLD, CLD, fp + fo_inb + 2 * CLD, vh, CLD, M3, CLD, CLD, 0, 1.f, 0);
    float* ctxf = alloc((size_t)M3 * CLD);
    fuse_attn_kernel<<<CB * CAH, 256, 0, stream>>>(qh, kh, vh, ctxf);
    float* mo = alloc((size_t)M3 * CLD);
    gemm(ctxf, CLD, fp + fo_outw, CLD, fp + fo_outb, mo, CLD, M3, CLD, CLD, 0, 1.f, 0);
    // per-position reduction linears (batched over s=0..2)
    float* redb = alloc((size_t)CB * 3 * CRED);
    gemm(mo, 3 * CLD, fp + fo_redw, CRED, fp + fo_redb, redb, 3 * CRED,
         CB, CRED, CLD, 0, 1.f, 0,
         3, 1,
         (long long)CLD, 0, (long long)CLD * CRED, 0, (long long)CRED, 0,
         (long long)CRED, 0);
    // hidden = [ld | pooled | fat2]
    float* hidden = alloc((size_t)CB * 1792);
    gemm(redb, 3 * CRED, fp + fo_ldw, CRED, fp + fo_ldb, hidden, 1792, CB, CRED, 3 * CRED, 0, 1.f, 0);
    copy2d(pooled, CD, hidden + 256, 1792, CB, CD);
    copy2d(fat2, CD, hidden + 1024, 1792, CB, CD);
    float* m1 = alloc((size_t)CB * 896);
    gemm(hidden, 1792, fp + fo_m1w, 896, fp + fo_m1b, m1, 896, CB, 896, 1792, 0, 1.f, 3 /*lrelu .01*/);
    float* logitsb = alloc((size_t)CB * 2);
    gemm(m1, 896, fp + fo_m2w, 2, fp + fo_m2b, logitsb, 2, CB, 2, 896, 0, 1.f, 0);

    // =================== outputs: (logits, fat2, pooled) ===================
    float* out = (float*)d_out;
    copy2d(logitsb, 2, out, 2, CB, 2);
    copy2d(fat2, CD, out + CB * 2, CD, CNF, CD);
    copy2d(pooled, CD, out + CB * 2 + CNF * CD, CD, CB, CD);
}
